// VanillaSpanBase_15479062134864
// MI455X (gfx1250) — matrix operand order repaired and checked by reading, not compiled, not run on hardware
//
#include <hip/hip_runtime.h>

#define BSZ 4
#define SEQ 192
#define HID 768
#define ATT 256
#define NCLS 9

typedef __attribute__((ext_vector_type(16))) __bf16 v16bf;
typedef __attribute__((ext_vector_type(8)))  __bf16 v8bf;
typedef __attribute__((ext_vector_type(8)))  float  v8f;
typedef unsigned int u32x4 __attribute__((ext_vector_type(4)));
typedef int          i32x8 __attribute__((ext_vector_type(8)));
typedef int          i32x4 __attribute__((ext_vector_type(4)));

// ---------- helpers ----------
__device__ __forceinline__ v8f splat8(float x) {
  v8f v = {x, x, x, x, x, x, x, x};
  return v;
}

__device__ __forceinline__ v16bf cat16(v8bf lo, v8bf hi) {
  return __builtin_shufflevector(lo, hi, 0, 1, 2, 3, 4, 5, 6, 7,
                                 8, 9, 10, 11, 12, 13, 14, 15);
}

// Packed bf16 max (2 bf16 per dword) — V_PK_MAX_NUM_BF16 (VOP3P op 44).
__device__ __forceinline__ unsigned pk_max_bf16(unsigned a, unsigned b) {
  unsigned d;
  asm("v_pk_max_num_bf16 %0, %1, %2" : "=v"(d) : "v"(a), "v"(b));
  return d;
}

// ---------- WMMA wrapper ----------
__device__ __forceinline__ v8f wmma_bf16(v16bf a, v16bf b, v8f c) {
  return __builtin_amdgcn_wmma_f32_16x16x32_bf16(false, a, false, b, (short)0, c,
                                                 false, false);
}

// A fragment (16x32 bf16, row-major, leading dim ld, K offset k0).
// A order used here (contiguous): lane m=L&15, g=L>>4; element h -> K=16g+h.
// Both 8-element runs contiguous => ds_load_b128.
__device__ __forceinline__ v16bf load_A_lds(const __bf16* base, int ld, int k0) {
  const int lane = threadIdx.x & 31;
  const int m = lane & 15, g = lane >> 4;
  const __bf16* p = base + m * ld + k0 + 16 * g;
  v8bf lo = *(const v8bf*)(p);
  v8bf hi = *(const v8bf*)(p + 8);
  return cat16(lo, hi);
}

// B fragment from pre-packed LDS: fragment = 32 lanes x 16 contiguous bf16.
__device__ __forceinline__ v16bf load_B_packed(const __bf16* frag_base) {
  const int lane = threadIdx.x & 31;
  const __bf16* p = frag_base + lane * 16;
  v8bf lo = *(const v8bf*)(p);
  v8bf hi = *(const v8bf*)(p + 8);
  return cat16(lo, hi);
}

// Pack a (nks*32) x (nnt*16) B matrix from global f32 into B-fragment layout:
// dst[(ks*nnt+nt)*512 + lane*16 + h] = B[ks*32 + 16g + h][nt*16 + n]
// B order used here (contiguous, matches A): lane n=L&15, g=L>>4; element h -> K = 16g+h.
__device__ __forceinline__ void pack_B(__bf16* dst, const float* src, int ld,
                                       int nks, int nnt, int ncols) {
  const int lane = threadIdx.x & 31;
  const int n = lane & 15, g = lane >> 4;
  for (int ks = 0; ks < nks; ++ks)
    for (int nt = 0; nt < nnt; ++nt) {
      __bf16* d = dst + ((ks * nnt + nt) * 32 + lane) * 16;
      const int col = nt * 16 + n;
#pragma unroll
      for (int h = 0; h < 16; ++h) {
        float v = (col < ncols) ? src[(ks * 32 + 16 * g + h) * ld + col] : 0.0f;
        d[h] = (__bf16)v;
      }
    }
}

// ======================= Kernel 1: proj (bf16) =======================
__global__ __launch_bounds__(32) void proj_kernel(
    const float* __restrict__ memory, const float* __restrict__ w_p1,
    const float* __restrict__ b_p1, const float* __restrict__ w_p2,
    const float* __restrict__ b_p2, __bf16* __restrict__ proj) {
  __shared__ __bf16 mem_tile[16 * HID];        // 24 KB, ld = 768
  __shared__ __bf16 hid_tile[16 * 264];        // ~8.3 KB, ld = 264 (padded)
  const int lane = threadIdx.x & 31;
  const int b = blockIdx.x / (SEQ / 16);
  const int t = blockIdx.x % (SEQ / 16);
  const int row0 = t * 16;

  const float* msrc = memory + (size_t)(b * SEQ + row0) * HID;
  for (int idx = lane; idx < 16 * HID; idx += 32) mem_tile[idx] = (__bf16)msrc[idx];
  __syncthreads();

  const int n = lane & 15, g = lane >> 4;

  // hidden = relu(mem @ w_p1 + b1)  (16 x 256, K = 768); 8 resident accumulators.
  for (int half = 0; half < 2; ++half) {
    v8f acc[8];
#pragma unroll
    for (int t8 = 0; t8 < 8; ++t8) acc[t8] = splat8(b_p1[(half * 8 + t8) * 16 + n]);
    for (int k = 0; k < HID / 32; ++k) {
      v16bf a = load_A_lds(mem_tile, HID, k * 32);
#pragma unroll
      for (int t8 = 0; t8 < 8; ++t8) {
        v16bf bb;
#pragma unroll
        for (int h = 0; h < 16; ++h)
          bb[h] = (__bf16)w_p1[(k * 32 + 16 * g + h) * ATT + (half * 8 + t8) * 16 + n];
        acc[t8] = wmma_bf16(a, bb, acc[t8]);
      }
    }
#pragma unroll
    for (int t8 = 0; t8 < 8; ++t8)
#pragma unroll
      for (int r = 0; r < 8; ++r)
        hid_tile[(r + 8 * g) * 264 + (half * 8 + t8) * 16 + n] =
            (__bf16)fmaxf(acc[t8][r], 0.0f);
  }
  __syncthreads();

  // proj = hidden @ w_p2 + b2  (16 x 256, K = 256; no relu)
  for (int half = 0; half < 2; ++half) {
    v8f acc[8];
#pragma unroll
    for (int t8 = 0; t8 < 8; ++t8) acc[t8] = splat8(b_p2[(half * 8 + t8) * 16 + n]);
    for (int k = 0; k < ATT / 32; ++k) {
      v16bf a = load_A_lds(hid_tile, 264, k * 32);
#pragma unroll
      for (int t8 = 0; t8 < 8; ++t8) {
        v16bf bb;
#pragma unroll
        for (int h = 0; h < 16; ++h)
          bb[h] = (__bf16)w_p2[(k * 32 + 16 * g + h) * ATT + (half * 8 + t8) * 16 + n];
        acc[t8] = wmma_bf16(a, bb, acc[t8]);
      }
    }
#pragma unroll
    for (int t8 = 0; t8 < 8; ++t8)
#pragma unroll
      for (int r = 0; r < 8; ++r)
        proj[(size_t)(b * SEQ + row0 + r + 8 * g) * ATT + (half * 8 + t8) * 16 + n] =
            (__bf16)acc[t8][r];
  }
}

// ============ Kernel 2: fused span-max + 4-layer MLP + mask ============
// One block (one wave) per (b, i). Proj rows for the NEXT chunk are prefetched
// into LDS by the Tensor Data Mover while WMMAs run on the current chunk.
struct Smem {                       // single object => LDS offsets well-defined
  __bf16 w1f[32 * 512];             // 32 KB: 8 Kstep x 4 Ntile B-fragments
  __bf16 w2f[8 * 512];
  __bf16 w3f[8 * 512];
  __bf16 w4f[2 * 512];
  __bf16 span_buf[16 * 264];        // 16 x 256 span tile, padded ld
  __bf16 hA[16 * 72];
  __bf16 hB[16 * 72];
  float bias1[64], bias2[64], bias3[64], bias4[16];
  __bf16 stage[2][16 * ATT];        // TDM double-buffer: 2 x 8 KB proj slabs
};

__global__ __launch_bounds__(32) void span_cls_kernel(
    const __bf16* __restrict__ proj, const int* __restrict__ embeds_length,
    const float* __restrict__ w_c1, const float* __restrict__ b_c1,
    const float* __restrict__ w_c2, const float* __restrict__ b_c2,
    const float* __restrict__ w_c3, const float* __restrict__ b_c3,
    const float* __restrict__ w_c4, const float* __restrict__ b_c4,
    float* __restrict__ out) {
  __shared__ Smem sm;

  const int lane = threadIdx.x & 31;
  const int b = blockIdx.x / SEQ;
  const int i = blockIdx.x % SEQ;
  const int len = embeds_length[b];

  pack_B(sm.w1f, w_c1, 64, 8, 4, 64);
  pack_B(sm.w2f, w_c2, 64, 2, 4, 64);
  pack_B(sm.w3f, w_c3, 64, 2, 4, 64);
  pack_B(sm.w4f, w_c4, NCLS, 2, 1, NCLS);
  for (int idx = lane; idx < 64; idx += 32) {
    sm.bias1[idx] = b_c1[idx]; sm.bias2[idx] = b_c2[idx]; sm.bias3[idx] = b_c3[idx];
  }
  if (lane < 16) sm.bias4[lane] = (lane < NCLS) ? b_c4[lane] : 0.0f;

  const int n = lane & 15, g = lane >> 4;
  const int nch = (SEQ - i + 15) >> 4;
  float* outrow = out + (size_t)(b * SEQ + i) * SEQ * NCLS;

  // ---- TDM prefetch of one 16x256 bf16 proj slab (contiguous 8 KB) ----
#if __has_builtin(__builtin_amdgcn_tensor_load_to_lds)
  #define HAVE_TDM 1
#else
  #define HAVE_TDM 0
#endif
  auto issue_chunk = [&](int c_next) {
    const int j0n = i + c_next * 16;
#if HAVE_TDM
    unsigned long long ga =
        (unsigned long long)(size_t)(proj + (size_t)(b * SEQ + j0n) * ATT);
    // Low 32 bits of a flat LDS pointer are the LDS byte offset.
    unsigned lds_off = (unsigned)(size_t)(&sm.stage[c_next & 1][0]);
    // D# group0: count=1 | lds_addr | global_addr | type=2.
    u32x4 g0 = {1u, lds_off, (unsigned)(ga & 0xFFFFFFFFu),
                (unsigned)(((ga >> 32) & 0x01FFFFFFu) | (2u << 30))};
    // D# group1: data_size=2B; 1-D tile: tile_dim0=4096 elems,
    // tensor_dim0=(SEQ-j0)*256 so overhang rows read as zero (only the final
    // chunk overhangs, where the carry is never reused).
    unsigned td0 = (unsigned)((SEQ - j0n) * ATT);
    i32x8 g1 = {(int)0x00010000u,
                (int)((td0 & 0xFFFFu) << 16),
                (int)(((td0 >> 16) & 0xFFFFu) | (1u << 16)),  // | tensor_dim1=1
                (int)(4096u << 16),                           // tile_dim0
                0,
                (int)ATT,                                     // dim0 stride (unused 1-D)
                0, 0};
    i32x4 z4 = {0, 0, 0, 0};
    i32x8 z8 = {0, 0, 0, 0, 0, 0, 0, 0};
    __builtin_amdgcn_tensor_load_to_lds(g0, g1, z4, z4, z8, 0);
#else
    unsigned* dst = (unsigned*)&sm.stage[c_next & 1][0];
    for (int idx = lane; idx < 16 * ATT / 2; idx += 32) {
      int jj = idx >> 7, d = idx & 127;
      int j = j0n + jj; if (j > SEQ - 1) j = SEQ - 1;
      dst[idx] = ((const unsigned*)(proj + (size_t)(b * SEQ + j) * ATT))[d];
    }
#endif
  };

  issue_chunk(0);

  // Running max carry: 4 dwords = 8 packed bf16 features/lane, init -inf.
  unsigned carry[4];
#pragma unroll
  for (int q = 0; q < 4; ++q) carry[q] = 0xFF80FF80u;

  __syncthreads();  // weights/biases visible

  for (int c = 0; c < nch; ++c) {
    const int j0 = i + c * 16;
    const int buf = c & 1;
    if (c + 1 < nch) {
      issue_chunk(c + 1);
#if HAVE_TDM
      __builtin_amdgcn_s_wait_tensorcnt(1);  // oldest (current chunk) complete
#endif
    } else {
#if HAVE_TDM
      __builtin_amdgcn_s_wait_tensorcnt(0);
#endif
    }
#if !HAVE_TDM
    __syncthreads();
#endif
    asm volatile("" ::: "memory");

    // Build 16 span rows from staged slab: packed-bf16 running max.
    for (int jj = 0; jj < 16; ++jj) {
      const unsigned* pr = (const unsigned*)(&sm.stage[buf][jj * ATT]);
      unsigned* sb = (unsigned*)(&sm.span_buf[jj * 264]);
#pragma unroll
      for (int q = 0; q < 4; ++q) {
        carry[q] = pk_max_bf16(carry[q], pr[lane + 32 * q]);
        sb[lane + 32 * q] = carry[q];
      }
    }
    __syncthreads();

    // c1: h = relu(span @ w_c1 + b_c1)   (K=256, N=64), A fragments resident.
    {
      v16bf a1[8];
#pragma unroll
      for (int k = 0; k < 8; ++k) a1[k] = load_A_lds(sm.span_buf, 264, k * 32);
#pragma unroll
      for (int nt = 0; nt < 4; ++nt) {
        v8f acc = splat8(sm.bias1[nt * 16 + n]);
#pragma unroll
        for (int k = 0; k < 8; ++k)
          acc = wmma_bf16(a1[k], load_B_packed(sm.w1f + (k * 4 + nt) * 512), acc);
#pragma unroll
        for (int r = 0; r < 8; ++r)
          sm.hA[(r + 8 * g) * 72 + nt * 16 + n] = (__bf16)fmaxf(acc[r], 0.0f);
      }
    }
    __syncthreads();

    // c2: h = relu(h @ w_c2 + b_c2)      (K=64, N=64)
    {
      v16bf a0 = load_A_lds(sm.hA, 72, 0);
      v16bf a1 = load_A_lds(sm.hA, 72, 32);
#pragma unroll
      for (int nt = 0; nt < 4; ++nt) {
        v8f acc = splat8(sm.bias2[nt * 16 + n]);
        acc = wmma_bf16(a0, load_B_packed(sm.w2f + (0 * 4 + nt) * 512), acc);
        acc = wmma_bf16(a1, load_B_packed(sm.w2f + (1 * 4 + nt) * 512), acc);
#pragma unroll
        for (int r = 0; r < 8; ++r)
          sm.hB[(r + 8 * g) * 72 + nt * 16 + n] = (__bf16)fmaxf(acc[r], 0.0f);
      }
    }
    __syncthreads();

    // c3: h = relu(h @ w_c3 + b_c3)      (K=64, N=64)
    {
      v16bf a0 = load_A_lds(sm.hB, 72, 0);
      v16bf a1 = load_A_lds(sm.hB, 72, 32);
#pragma unroll
      for (int nt = 0; nt < 4; ++nt) {
        v8f acc = splat8(sm.bias3[nt * 16 + n]);
        acc = wmma_bf16(a0, load_B_packed(sm.w3f + (0 * 4 + nt) * 512), acc);
        acc = wmma_bf16(a1, load_B_packed(sm.w3f + (1 * 4 + nt) * 512), acc);
#pragma unroll
        for (int r = 0; r < 8; ++r)
          sm.hA[(r + 8 * g) * 72 + nt * 16 + n] = (__bf16)fmaxf(acc[r], 0.0f);
      }
    }
    __syncthreads();

    // c4: logits = h @ w_c4 + b_c4       (K=64, N=9 padded to 16)
    {
      v16bf a0 = load_A_lds(sm.hA, 72, 0);
      v16bf a1 = load_A_lds(sm.hA, 72, 32);
      v8f acc = splat8(sm.bias4[n]);
      acc = wmma_bf16(a0, load_B_packed(sm.w4f + 0 * 512), acc);
      acc = wmma_bf16(a1, load_B_packed(sm.w4f + 1 * 512), acc);
      if (n < NCLS) {
#pragma unroll
        for (int r = 0; r < 8; ++r) {
          const int j = j0 + r + 8 * g;
          if (j < SEQ) outrow[j * NCLS + n] = (j < len) ? acc[r] : 0.0f;
        }
      }
    }
    __syncthreads();
  }

  // Zero-fill the j < i (lower-triangle) region this block owns.
  for (int idx = lane; idx < i * NCLS; idx += 32) outrow[idx] = 0.0f;
}

extern "C" void kernel_launch(void* const* d_in, const int* in_sizes, int n_in,
                              void* d_out, int out_size, void* d_ws, size_t ws_size,
                              hipStream_t stream) {
  (void)in_sizes; (void)n_in; (void)out_size; (void)ws_size;
  const float* memory        = (const float*)d_in[0];
  const int*   embeds_length = (const int*)  d_in[1];
  const float* w_p1 = (const float*)d_in[2];
  const float* b_p1 = (const float*)d_in[3];
  const float* w_p2 = (const float*)d_in[4];
  const float* b_p2 = (const float*)d_in[5];
  const float* w_c1 = (const float*)d_in[6];
  const float* b_c1 = (const float*)d_in[7];
  const float* w_c2 = (const float*)d_in[8];
  const float* b_c2 = (const float*)d_in[9];
  const float* w_c3 = (const float*)d_in[10];
  const float* b_c3 = (const float*)d_in[11];
  const float* w_c4 = (const float*)d_in[12];
  const float* b_c4 = (const float*)d_in[13];
  float*  out  = (float*)d_out;
  __bf16* proj = (__bf16*)d_ws;  // BSZ*SEQ*ATT bf16 = 384 KB scratch

  proj_kernel<<<BSZ * (SEQ / 16), 32, 0, stream>>>(memory, w_p1, b_p1, w_p2, b_p2, proj);
  span_cls_kernel<<<BSZ * SEQ, 32, 0, stream>>>(proj, embeds_length,
                                                w_c1, b_c1, w_c2, b_c2,
                                                w_c3, b_c3, w_c4, b_c4, out);
}
